// MultiHeadAttention_5841155523186
// MI455X (gfx1250) — compile-verified, matrix operand order checked by reading, not run on hardware
//
#include <hip/hip_runtime.h>

// ---------------------------------------------------------------------------
// Multi-head attention for MI455X (gfx1250): bf16 WMMA + double-buffered TDM.
// B=2, N=2048, D=1024, H=16, dh=64.
// d_out = [ out (2*2048*1024 f32) | dot_prod (32*2048*2048 f32) ]
// ---------------------------------------------------------------------------

#define DEVI __device__ __forceinline__

typedef __attribute__((ext_vector_type(16))) __bf16 v16bf;
typedef __attribute__((ext_vector_type(8)))  float  v8f;
typedef __attribute__((ext_vector_type(4)))  unsigned int v4u;
typedef __attribute__((ext_vector_type(8)))  int v8i;
typedef __attribute__((ext_vector_type(4)))  int v4i;
typedef unsigned short u16t;
typedef unsigned int   u32t;

static constexpr int kB  = 2;
static constexpr int kN  = 2048;
static constexpr int kD  = 1024;
static constexpr int kH  = 16;
static constexpr int kDh = 64;
static constexpr int kM  = kB * kN;           // 4096 total tokens
static constexpr float kNegInf = -1e20f;

union FragU { uint4 q[2]; v16bf v; };

DEVI u16t f2bf(float f) {
  union { float f; u32t u; } x; x.f = f;
  u32t r = x.u + 0x7FFFu + ((x.u >> 16) & 1u);   // round-to-nearest-even
  return (u16t)(r >> 16);
}

DEVI unsigned ldsoff(const void* p) {            // LDS byte offset of a shared ptr
  return (unsigned)(unsigned long long)p;
}

// Load a 16x32 bf16 WMMA fragment (A or B operand) where each lane's
// 32-element K run lives contiguously at row (lane&15) with leading dim ld.
// Layout per CDNA5 ISA 7.12.2 (16-bit A 16x32):
//   lanes 0-15 : VGPR0-3 = K0..7,  VGPR4-7 = K16..23
//   lanes 16-31: VGPR0-3 = K8..15, VGPR4-7 = K24..31
DEVI v16bf load_frag(const u16t* __restrict__ base, int ld, int lane) {
  const int hi = (lane >> 4) & 1;
  const u16t* p = base + (size_t)(lane & 15) * ld + hi * 8;
  FragU f;
  f.q[0] = *(const uint4*)(p);        // 8 bf16: K 0..7  (or 8..15)
  f.q[1] = *(const uint4*)(p + 16);   // 8 bf16: K 16..23 (or 24..31)
  return f.v;
}

DEVI v8f wmma_bf16(v16bf a, v16bf b, v8f c) {
  return __builtin_amdgcn_wmma_f32_16x16x32_bf16(
      false, a, false, b, (short)0, c, false, false);
}

// ---------------------------------------------------------------------------
// Tensor Data Mover: 2D tile (tile_w x tile_h elements of 2 bytes) from global
// row-major (stride elems) into LDS, rows compacted. D# per ISA ch.8.
// ---------------------------------------------------------------------------
#if __has_builtin(__builtin_amdgcn_tensor_load_to_lds)
#define HAVE_TDM 1
DEVI void tdm_load_2d_bf16(unsigned lds_off, const void* gsrc,
                           unsigned tile_w, unsigned tile_h,
                           unsigned tensor_w, unsigned tensor_h,
                           unsigned stride_elems) {
  unsigned long long ga = (unsigned long long)gsrc;
  v4u g0;
  g0.x = 1u;                                              // count=1, user mode
  g0.y = lds_off;                                         // lds_addr (bytes)
  g0.z = (unsigned)(ga & 0xFFFFFFFFu);                    // global_addr[31:0]
  g0.w = (unsigned)((ga >> 32) & 0x01FFFFFFu) | (2u << 30); // [56:32] | type=2
  unsigned w0 = (1u << 16);                               // data_size=1 (2B)
  unsigned w1 = (tensor_w & 0xFFFFu) << 16;               // tensor_dim0[15:0]
  unsigned w2 = ((tensor_w >> 16) & 0xFFFFu) | ((tensor_h & 0xFFFFu) << 16);
  unsigned w3 = ((tensor_h >> 16) & 0xFFFFu) | ((tile_w & 0xFFFFu) << 16);
  unsigned w4 = (tile_h & 0xFFFFu);                       // tile_dim1 (tile_dim2=0)
  unsigned w5 = stride_elems;                             // dim0 stride lo32
  unsigned w6 = 0, w7 = 0;
  v8i g1 = (v8i){(int)w0,(int)w1,(int)w2,(int)w3,(int)w4,(int)w5,(int)w6,(int)w7};
  v4i z4 = (v4i){0, 0, 0, 0};
#if defined(__clang_major__) && (__clang_major__ >= 23)
  v8i z8 = (v8i){0, 0, 0, 0, 0, 0, 0, 0};
  __builtin_amdgcn_tensor_load_to_lds(g0, g1, z4, z4, z8, 0);
#else
  __builtin_amdgcn_tensor_load_to_lds(g0, g1, z4, z4, 0);
#endif
}
#endif

// ---------------------------------------------------------------------------
// fp32 -> bf16 elementwise conversion
// ---------------------------------------------------------------------------
__global__ void k_cvt_bf16(const float* __restrict__ src, u16t* __restrict__ dst, int n) {
  int i = blockIdx.x * blockDim.x + threadIdx.x;
  if (i < n) dst[i] = f2bf(src[i]);
}

// ---------------------------------------------------------------------------
// Fused QKV projection: C = X @ W^T + bias.  One wave per 16x16 output tile,
// 3 WMMAs per K-step sharing the A fragment. Q pre-scaled by 1/sqrt(dh).
// Q,K stored [B,H,N,dh] bf16; V stored transposed [B,H,dh,N] bf16.
// ---------------------------------------------------------------------------
__global__ __launch_bounds__(32) void k_qkv(
    const u16t* __restrict__ X,
    const u16t* __restrict__ Wq, const u16t* __restrict__ Wk, const u16t* __restrict__ Wv,
    const float* __restrict__ bq, const float* __restrict__ bk, const float* __restrict__ bv,
    u16t* __restrict__ Qb, u16t* __restrict__ Kb, u16t* __restrict__ Vt)
{
  const int lane = threadIdx.x;
  const int nt = blockIdx.x & (kD / 16 - 1);   // 64 col tiles
  const int mt = blockIdx.x >> 6;              // 256 row tiles

  const u16t* ab = X  + (size_t)mt * 16 * kD;
  const u16t* qb = Wq + (size_t)nt * 16 * kD;
  const u16t* kb = Wk + (size_t)nt * 16 * kD;
  const u16t* vb = Wv + (size_t)nt * 16 * kD;

  v8f cq = {}, ck = {}, cv = {};
  for (int k = 0; k < kD; k += 32) {
    v16bf a  = load_frag(ab + k, kD, lane);
    v16bf b0 = load_frag(qb + k, kD, lane);
    v16bf b1 = load_frag(kb + k, kD, lane);
    v16bf b2 = load_frag(vb + k, kD, lane);
    cq = wmma_bf16(a, b0, cq);
    ck = wmma_bf16(a, b1, ck);
    cv = wmma_bf16(a, b2, cv);
  }

  const int hi  = (lane >> 4) & 1;
  const int col = nt * 16 + (lane & 15);       // global D column
  const int h   = col >> 6;                    // head
  const int dc  = col & (kDh - 1);             // dh column
  const float vq = bq[col], vk = bk[col], vv = bv[col];

#pragma unroll
  for (int r = 0; r < 8; ++r) {
    const int tok = mt * 16 + r + 8 * hi;
    const int b   = tok >> 11;                 // / 2048
    const int n   = tok & (kN - 1);
    const int bh  = b * kH + h;
    Qb[((size_t)bh * kN + n) * kDh + dc] = f2bf((cq[r] + vq) * 0.125f); // /sqrt(64)
    Kb[((size_t)bh * kN + n) * kDh + dc] = f2bf(ck[r] + vk);
    Vt[((size_t)bh * kDh + dc) * kN + n] = f2bf(cv[r] + vv);
  }
}

// ---------------------------------------------------------------------------
// Scores: S = Q @ K^T (per head), masked -> fp32 dot_prod.
// Block = 4 waves sharing one (b,h): each wave owns one q tile (A fragments
// loaded once, reused for 32 kt tiles).  Each 16(token) x 64(dh) K tile is
// staged in LDS via double-buffered TDM (s_wait_tensorcnt(1) steady state),
// overlapping the DMA with the previous tile's WMMAs.
// ---------------------------------------------------------------------------
__global__ __launch_bounds__(128) void k_scores(
    const u16t* __restrict__ Qb, const u16t* __restrict__ Kb,
    const int* __restrict__ mask, float* __restrict__ dotp)
{
  __shared__ u16t sK[2][16 * kDh];             // two 2KB K tiles

  const int tid  = threadIdx.x;
  const int lane = tid & 31;
  const int wid  = tid >> 5;                   // 0..3
  const int ktg = blockIdx.x & 3;              // 4 groups of 32 kt tiles
  const int qtg = (blockIdx.x >> 2) & 31;      // 32 groups of 4 q tiles
  const int bh  = blockIdx.x >> 7;             // 32 (b,h)
  const int b   = bh >> 4;
  const int qt  = qtg * 4 + wid;

  const u16t* qbase = Qb + ((size_t)bh * kN + qt * 16) * kDh;
  const v16bf a0 = load_frag(qbase,      kDh, lane);
  const v16bf a1 = load_frag(qbase + 32, kDh, lane);

  const u16t* kbh = Kb + (size_t)bh * kN * kDh;
  const int kt0 = ktg * 32;
  const int nsteps = 32;

  const int hi   = (lane >> 4) & 1;
  const int ncol = lane & 15;

#ifdef HAVE_TDM
  if (wid == 0) {   // prime both buffers
    tdm_load_2d_bf16(ldsoff(&sK[0][0]), kbh + (size_t)(kt0 + 0) * 16 * kDh,
                     kDh, 16, kDh, kN, kDh);
    tdm_load_2d_bf16(ldsoff(&sK[1][0]), kbh + (size_t)(kt0 + 1) * 16 * kDh,
                     kDh, 16, kDh, kN, kDh);
  }
#endif

  for (int step = 0; step < nsteps; ++step) {
    const int kt  = kt0 + step;
    const int cur = step & 1;

#ifdef HAVE_TDM
    if (wid == 0) {
      if (step + 1 < nsteps) __builtin_amdgcn_s_wait_tensorcnt(1);
      else                   __builtin_amdgcn_s_wait_tensorcnt(0);
    }
    __syncthreads();
#else
    for (int i = tid; i < (16 * kDh) / 8; i += 128) {   // 128 uint4 chunks
      ((uint4*)&sK[cur][0])[i] =
          *(const uint4*)(kbh + (size_t)kt * 16 * kDh + i * 8);
    }
    __syncthreads();
#endif

    v16bf b0 = load_frag(&sK[cur][0],  kDh, lane);      // ds_load_b128
    v16bf b1 = load_frag(&sK[cur][32], kDh, lane);
    v8f c = {};
    c = wmma_bf16(a0, b0, c);
    c = wmma_bf16(a1, b1, c);

    const int kk = kt * 16 + ncol;
    const bool dead = (mask[b * kN + kk] == 0);
    float* outp = dotp + ((size_t)bh * kN + qt * 16) * kN + kk;
#pragma unroll
    for (int r = 0; r < 8; ++r)
      outp[(size_t)(r + 8 * hi) * kN] = dead ? kNegInf : c[r];

    __syncthreads();                                    // done reading sK[cur]
#ifdef HAVE_TDM
    if (wid == 0 && step + 2 < nsteps)                  // refill sK[cur]
      tdm_load_2d_bf16(ldsoff(&sK[cur][0]),
                       kbh + (size_t)(kt + 2) * 16 * kDh, kDh, 16, kDh, kN, kDh);
#endif
  }
}

// ---------------------------------------------------------------------------
// Softmax row statistics: row max and 1/sum(exp). One wave per row.
// ---------------------------------------------------------------------------
__global__ __launch_bounds__(32) void k_stats(
    const float* __restrict__ dotp, float* __restrict__ rmax, float* __restrict__ rinv)
{
  const int lane = threadIdx.x;
  const size_t row = blockIdx.x;
  const float* p = dotp + row * kN;

  float m = -3.0e38f;
  for (int i = lane; i < kN; i += 32) m = fmaxf(m, p[i]);
#pragma unroll
  for (int o = 16; o; o >>= 1) m = fmaxf(m, __shfl_xor(m, o, 32));

  float s = 0.f;
  for (int i = lane; i < kN; i += 32) s += __expf(p[i] - m);
#pragma unroll
  for (int o = 16; o; o >>= 1) s += __shfl_xor(s, o, 32);

  if (lane == 0) { rmax[row] = m; rinv[row] = 1.0f / s; }
}

// ---------------------------------------------------------------------------
// ctx = softmax(S) @ V.  Block = 4 waves sharing one (b,h); each wave owns one
// q tile.  The 64(dh) x 32(k) transposed-V tile is staged in LDS via
// double-buffered TDM; bf16 P fragments built on the fly from fp32 scores.
// Output ctx bf16 in merged-head layout [B,N,H*dh] = [M,D].
// ---------------------------------------------------------------------------
__global__ __launch_bounds__(128) void k_pv(
    const float* __restrict__ dotp,
    const float* __restrict__ rmax, const float* __restrict__ rinv,
    const u16t* __restrict__ Vt, u16t* __restrict__ Ctx)
{
  __shared__ u16t sV[2][kDh * 32];             // two 4KB V tiles (64 x 32)

  const int tid  = threadIdx.x;
  const int lane = tid & 31;
  const int wid  = tid >> 5;                   // 0..3
  const int qtg = blockIdx.x & 31;             // 32 groups of 4 q tiles
  const int bh  = blockIdx.x >> 5;             // 32 (b,h)
  const int qt  = qtg * 4 + wid;

  const int hi = (lane >> 4) & 1;
  const int mr = lane & 15;

  const size_t rowi = (size_t)bh * kN + qt * 16 + mr;
  const float* srow = dotp + rowi * kN;
  const float mx  = rmax[rowi];
  const float inv = rinv[rowi];

  const u16t* vbh = Vt + (size_t)bh * kDh * kN;   // [dh=64, N=2048] row-major
  const int nsteps = kN / 32;                     // 64

  v8f acc[4] = {{}, {}, {}, {}};

#ifdef HAVE_TDM
  if (wid == 0) {   // prime both buffers
    tdm_load_2d_bf16(ldsoff(&sV[0][0]), vbh + 0,  32, kDh, kN, kDh, kN);
    tdm_load_2d_bf16(ldsoff(&sV[1][0]), vbh + 32, 32, kDh, kN, kDh, kN);
  }
#endif

  for (int step = 0; step < nsteps; ++step) {
    const int kk  = step * 32;
    const int cur = step & 1;

#ifdef HAVE_TDM
    if (wid == 0) {
      if (step + 1 < nsteps) __builtin_amdgcn_s_wait_tensorcnt(1);
      else                   __builtin_amdgcn_s_wait_tensorcnt(0);
    }
    __syncthreads();
#else
    for (int i = tid; i < (kDh * 32) / 8; i += 128) {   // 256 uint4 chunks
      const int row = i >> 2, c4 = i & 3;
      ((uint4*)&sV[cur][0])[i] = *(const uint4*)(vbh + (size_t)row * kN + kk + c4 * 8);
    }
    __syncthreads();
#endif

    // ---- build P fragment (16x32 bf16) from fp32 scores for this lane's row
    const int ks = kk + hi * 8;
    u32t w[8];
#pragma unroll
    for (int i = 0; i < 4; ++i) {
      float2 s0 = *(const float2*)(srow + ks + 2 * i);        // K 0..7 / 8..15
      float2 s1 = *(const float2*)(srow + ks + 16 + 2 * i);   // K 16..23 / 24..31
      float p0 = __expf(s0.x - mx) * inv;
      float p1 = __expf(s0.y - mx) * inv;
      float p2 = __expf(s1.x - mx) * inv;
      float p3 = __expf(s1.y - mx) * inv;
      w[i]     = (u32t)f2bf(p0) | ((u32t)f2bf(p1) << 16);
      w[4 + i] = (u32t)f2bf(p2) | ((u32t)f2bf(p3) << 16);
    }
    FragU fa;
    fa.q[0] = make_uint4(w[0], w[1], w[2], w[3]);
    fa.q[1] = make_uint4(w[4], w[5], w[6], w[7]);
    const v16bf a = fa.v;

    // ---- 4 WMMAs against LDS-resident transposed-V fragments ----
#pragma unroll
    for (int j = 0; j < 4; ++j) {
      v16bf bj = load_frag(&sV[cur][(size_t)j * 16 * 32], 32, lane); // ds_load_b128
      acc[j] = wmma_bf16(a, bj, acc[j]);
    }

    __syncthreads();                                    // done reading sV[cur]
#ifdef HAVE_TDM
    if (wid == 0 && step + 2 < nsteps)                  // refill sV[cur]
      tdm_load_2d_bf16(ldsoff(&sV[cur][0]), vbh + kk + 64, 32, kDh, kN, kDh, kN);
#endif
  }

  const int bb = bh >> 4, h = bh & 15;
#pragma unroll
  for (int j = 0; j < 4; ++j) {
    const int col = h * kDh + j * 16 + mr;
#pragma unroll
    for (int r = 0; r < 8; ++r) {
      const int tok = qt * 16 + r + 8 * hi;
      Ctx[((size_t)bb * kN + tok) * kD + col] = f2bf(acc[j][r]);
    }
  }
}

// ---------------------------------------------------------------------------
// Output projection: out = ctx @ Wo^T + bo (fp32).  One wave per 16x64 strip:
// A fragment reused across 4 B fragments.
// ---------------------------------------------------------------------------
__global__ __launch_bounds__(32) void k_oproj(
    const u16t* __restrict__ Ctx, const u16t* __restrict__ Wo,
    const float* __restrict__ bo, float* __restrict__ out)
{
  const int lane = threadIdx.x;
  const int ntg = blockIdx.x & 15;             // 16 groups of 4 col tiles
  const int mt  = blockIdx.x >> 4;             // 256 row tiles

  const u16t* ab = Ctx + (size_t)mt * 16 * kD;

  v8f acc[4] = {{}, {}, {}, {}};
  for (int k = 0; k < kD; k += 32) {
    v16bf a = load_frag(ab + k, kD, lane);
#pragma unroll
    for (int j = 0; j < 4; ++j) {
      const u16t* bb = Wo + (size_t)(ntg * 4 + j) * 16 * kD;
      v16bf b = load_frag(bb + k, kD, lane);
      acc[j] = wmma_bf16(a, b, acc[j]);
    }
  }

  const int hi = (lane >> 4) & 1;
#pragma unroll
  for (int j = 0; j < 4; ++j) {
    const int col = (ntg * 4 + j) * 16 + (lane & 15);
    const float bias = bo[col];
#pragma unroll
    for (int r = 0; r < 8; ++r)
      out[((size_t)mt * 16 + r + 8 * hi) * kD + col] = acc[j][r] + bias;
  }
}

// ---------------------------------------------------------------------------
// Host-side launcher
// ---------------------------------------------------------------------------
extern "C" void kernel_launch(void* const* d_in, const int* in_sizes, int n_in,
                              void* d_out, int out_size, void* d_ws, size_t ws_size,
                              hipStream_t stream) {
  const float* query = (const float*)d_in[0];
  const int*   mask  = (const int*)  d_in[1];
  const float* wq    = (const float*)d_in[2];
  const float* bq    = (const float*)d_in[3];
  const float* wk    = (const float*)d_in[4];
  const float* bk    = (const float*)d_in[5];
  const float* wv    = (const float*)d_in[6];
  const float* bv    = (const float*)d_in[7];
  const float* wo    = (const float*)d_in[8];
  const float* bo    = (const float*)d_in[9];

  float* out  = (float*)d_out;                       // [B,N,D]
  float* dotp = (float*)d_out + (size_t)kM * kD;     // [B*H, N, N]

  // Workspace carving (bf16 buffers + softmax stats). ~48.5 MB.
  char* ws = (char*)d_ws;
  size_t o = 0;
  u16t* Xbf = (u16t*)(ws + o); o += (size_t)kM * kD * 2;        // 8 MB
  u16t* Wqb = (u16t*)(ws + o); o += (size_t)kD * kD * 2;        // 2 MB
  u16t* Wkb = (u16t*)(ws + o); o += (size_t)kD * kD * 2;
  u16t* Wvb = (u16t*)(ws + o); o += (size_t)kD * kD * 2;
  u16t* Wob = (u16t*)(ws + o); o += (size_t)kD * kD * 2;
  u16t* Qb  = (u16t*)(ws + o); o += (size_t)kM * kD * 2;        // [B,H,N,dh]
  u16t* Kb  = (u16t*)(ws + o); o += (size_t)kM * kD * 2;        // [B,H,N,dh]
  u16t* Vt  = (u16t*)(ws + o); o += (size_t)kM * kD * 2;        // [B,H,dh,N]
  u16t* Ctx = (u16t*)(ws + o); o += (size_t)kM * kD * 2;        // [M,D]
  float* rmax = (float*)(ws + o); o += (size_t)kB * kH * kN * 4;
  float* rinv = (float*)(ws + o); o += (size_t)kB * kH * kN * 4;
  (void)ws_size; (void)n_in; (void)in_sizes; (void)out_size;

  // 1) fp32 -> bf16 conversions
  {
    const int nx = kM * kD, nw = kD * kD, thr = 256;
    k_cvt_bf16<<<(nx + thr - 1) / thr, thr, 0, stream>>>(query, Xbf, nx);
    k_cvt_bf16<<<(nw + thr - 1) / thr, thr, 0, stream>>>(wq, Wqb, nw);
    k_cvt_bf16<<<(nw + thr - 1) / thr, thr, 0, stream>>>(wk, Wkb, nw);
    k_cvt_bf16<<<(nw + thr - 1) / thr, thr, 0, stream>>>(wv, Wvb, nw);
    k_cvt_bf16<<<(nw + thr - 1) / thr, thr, 0, stream>>>(wo, Wob, nw);
  }

  // 2) fused QKV projection: (M/16)*(D/16) = 16384 waves
  k_qkv<<<dim3((kM / 16) * (kD / 16)), dim3(32), 0, stream>>>(
      Xbf, Wqb, Wkb, Wvb, bq, bk, bv, Qb, Kb, Vt);

  // 3) scores + mask -> dot_prod: 32 bh * 32 qt-groups * 4 kt-groups blocks
  k_scores<<<dim3(kB * kH * 32 * 4), dim3(128), 0, stream>>>(
      Qb, Kb, mask, dotp);

  // 4) softmax row stats: one wave per row (65536 rows)
  k_stats<<<dim3(kB * kH * kN), dim3(32), 0, stream>>>(dotp, rmax, rinv);

  // 5) PV with double-buffered TDM staging: 32 bh * 32 qt-groups blocks
  k_pv<<<dim3(kB * kH * (kN / 16 / 4)), dim3(128), 0, stream>>>(
      dotp, rmax, rinv, Vt, Ctx);

  // 6) output projection: 256 row tiles * 16 col groups
  k_oproj<<<dim3((kM / 16) * (kD / 16 / 4)), dim3(32), 0, stream>>>(
      Ctx, Wob, bo, out);
}
